// HeterogeneousLIVABLEPygModel_48430051230485
// MI455X (gfx1250) — matrix operand order repaired and checked by reading, not compiled, not run on hardware
//
#include <hip/hip_runtime.h>
#include <cmath>

#define N_NODES 50000
#define N_EDGES 320000
#define DIN     768
#define DMODEL  256
#define RREL    4
#define LLAYERS 3
#define BGRAPH  64
#define TSEQ    6
#define SDIM    128
#define SHID    256
#define NCLS    14
#define ALPHA_C 0.15f

typedef __attribute__((ext_vector_type(16))) __bf16 v16bf;
typedef __attribute__((ext_vector_type(8)))  __bf16 v8bf;
typedef __attribute__((ext_vector_type(8)))  float  v8f;

// fp32 -> bf16 round-to-nearest-even (one-shot converts / epilogues only)
__device__ __forceinline__ __bf16 f2bf(float f) {
  unsigned u = __builtin_bit_cast(unsigned, f);
  u += 0x7FFFu + ((u >> 16) & 1u);
  unsigned short s = (unsigned short)(u >> 16);
  return __builtin_bit_cast(__bf16, s);
}

__device__ __forceinline__ v16bf cat8(v8bf a, v8bf b) {
  return __builtin_shufflevector(a, b, 0,1,2,3,4,5,6,7,8,9,10,11,12,13,14,15);
}

__global__ void f32_to_bf16_kernel(const float* __restrict__ in, __bf16* __restrict__ out,
                                   size_t n) {
  size_t i = (size_t)blockIdx.x * blockDim.x + threadIdx.x;
  if (i < n) out[i] = f2bf(in[i]);
}

// ---------------------------------------------------------------------------
// C = act(A[M,K] @ W[N,K]^T + bias), bf16 operands, fp32 accumulate.
// One wave = 16x32 output tile (2 accumulators share one A fragment), with
// software-pipelined fragment loads (next k-chunk issued before current WMMAs).
// Optional fp32 output Cf and/or bf16 mirror Cb. N%32==0, M%16==0, K%32==0.
// ---------------------------------------------------------------------------
__global__ __launch_bounds__(256)
void gemm_bf16_wmma(const __bf16* __restrict__ A, int lda,
                    const __bf16* __restrict__ W, int ldw,
                    const float* __restrict__ bias,
                    float* __restrict__ Cf, __bf16* __restrict__ Cb, int ldc,
                    int M, int N, int K, int act)
{
  const int lane    = threadIdx.x & 31;
  const int n2tiles = N >> 5;
  const int mtiles  = M >> 4;
  const int wid     = blockIdx.x * (blockDim.x >> 5) + (threadIdx.x >> 5);
  if (wid >= mtiles * n2tiles) return;           // wave-uniform guard (EXEC stays full)
  const int mt  = wid / n2tiles;
  const int nt2 = wid - mt * n2tiles;            // columns [nt2*32, nt2*32+32)
  const int lh  = lane >> 4;                     // lane half (0/1)
  const int lr  = lane & 15;

  const __bf16* arow  = A + (size_t)(mt * 16 + lr) * lda;   // A row for this lane
  const __bf16* wrow0 = W + (size_t)(nt2 * 32 + lr) * ldw;  // output col = W row
  const __bf16* wrow1 = wrow0 + (size_t)16 * ldw;

  v8f acc0 = {}, acc1 = {};
  // ---- prologue: load k0 = 0 fragments ----
  // A fragment in contiguous k order (this code's choice): lanes 0-15 hold K 0..15, lanes 16-31 K 16..31
  const __bf16* pa = arow + lh * 16;
  v16bf acur = cat8(*(const v8bf*)pa, *(const v8bf*)(pa + 8));
  // B fragment, same contiguous k order as A: col = lane&15, lanes 0-15 hold K 0..15, lanes 16-31 K 16..31
  v16bf b0c = *(const v16bf*)(wrow0 + lh * 16);
  v16bf b1c = *(const v16bf*)(wrow1 + lh * 16);

  for (int k0 = 32; k0 < K; k0 += 32) {
    const __bf16* pn = arow + k0 + lh * 16;
    __builtin_prefetch((const void*)(pn + 32), 0, 3);
    v16bf anext = cat8(*(const v8bf*)pn, *(const v8bf*)(pn + 8));
    v16bf b0n = *(const v16bf*)(wrow0 + k0 + lh * 16);
    v16bf b1n = *(const v16bf*)(wrow1 + k0 + lh * 16);
    acc0 = __builtin_amdgcn_wmma_f32_16x16x32_bf16(false, acur, false, b0c,
                                                   (short)0, acc0, false, false);
    acc1 = __builtin_amdgcn_wmma_f32_16x16x32_bf16(false, acur, false, b1c,
                                                   (short)0, acc1, false, false);
    acur = anext; b0c = b0n; b1c = b1n;
  }
  acc0 = __builtin_amdgcn_wmma_f32_16x16x32_bf16(false, acur, false, b0c,
                                                 (short)0, acc0, false, false);
  acc1 = __builtin_amdgcn_wmma_f32_16x16x32_bf16(false, acur, false, b1c,
                                                 (short)0, acc1, false, false);

  const int col0 = nt2 * 32 + lr;
  const int col1 = col0 + 16;
  const float bv0 = bias ? bias[col0] : 0.0f;
  const float bv1 = bias ? bias[col1] : 0.0f;
#pragma unroll
  for (int v = 0; v < 8; ++v) {
    const size_t row = (size_t)(mt * 16 + v + 8 * lh);
    float r0 = acc0[v] + bv0;
    float r1 = acc1[v] + bv1;
    if (act == 1) { r0 = fmaxf(r0, 0.0f); r1 = fmaxf(r1, 0.0f); }
    if (Cf) { Cf[row * ldc + col0] = r0; Cf[row * ldc + col1] = r1; }
    if (Cb) { Cb[row * ldc + col0] = f2bf(r0); Cb[row * ldc + col1] = f2bf(r1); }
  }
}

// ---------------------------------------------------------------------------
// Fused edge-attention score, 16 edges x one relation per wave:
//   hidden = relu([h[dst]||h[src]] @ aW1[r]^T + ab1[r])   (16x128, 8 accumulators)
//   s = leaky_relu(hidden @ aW2[r] + ab2[r], 0.2) + mean(emb[r])
// All 8 B fragments loaded per k-chunk, then 8 back-to-back WMMAs (one wait
// per 8 matrix ops). cat never materialized: A gathered straight from bf16 h.
// ---------------------------------------------------------------------------
__global__ __launch_bounds__(256)
void edge_score_wmma(const __bf16* __restrict__ h,
                     const int* __restrict__ src, const int* __restrict__ dst,
                     const __bf16* __restrict__ aW1,  // [R,128,512] (this layer), bf16
                     const float* __restrict__ ab1,   // [R,128]
                     const float* __restrict__ aW2,   // [R,128]
                     const float* __restrict__ ab2,   // [R]
                     const float* __restrict__ embm,  // [R] mean(emb)
                     float* __restrict__ s4)          // [E,4]
{
  const int lane = threadIdx.x & 31;
  const int gw   = blockIdx.x * (blockDim.x >> 5) + (threadIdx.x >> 5);
  const int r    = gw & 3;
  const int tile = gw >> 2;                 // 16-edge tile
  const int lh   = lane >> 4, lr = lane & 15;
  const int e    = tile * 16 + lr;
  const int de   = dst[e];
  const int se   = src[e];
  const __bf16* hd = h + (size_t)de * DMODEL;
  const __bf16* hs = h + (size_t)se * DMODEL;
  const __bf16* w1 = aW1 + (size_t)r * 128 * 512;

  v8f acc[8];
  v8f zero = {};
#pragma unroll
  for (int i = 0; i < 8; ++i) acc[i] = zero;

  for (int k0 = 0; k0 < 512; k0 += 32) {
    const int ka = k0 + lh * 16;            // 8-elem run, never straddles the 256 split
    const int kb = ka + 8;
    v8bf a0 = *(const v8bf*)((ka < 256) ? (hd + ka) : (hs + (ka - 256)));
    v8bf a1 = *(const v8bf*)((kb < 256) ? (hd + kb) : (hs + (kb - 256)));
    v16bf afrag = cat8(a0, a1);
    v16bf bfrag[8];
#pragma unroll
    for (int nt = 0; nt < 8; ++nt)
      bfrag[nt] = *(const v16bf*)(w1 + (size_t)(nt * 16 + lr) * 512 + k0 + lh * 16);
#pragma unroll
    for (int nt = 0; nt < 8; ++nt)
      acc[nt] = __builtin_amdgcn_wmma_f32_16x16x32_bf16(false, afrag, false, bfrag[nt],
                                                        (short)0, acc[nt], false, false);
  }
  // second attention layer fused in-register
  float part[8];
#pragma unroll
  for (int v = 0; v < 8; ++v) part[v] = 0.0f;
#pragma unroll
  for (int nt = 0; nt < 8; ++nt) {
    const int   col = nt * 16 + lr;
    const float b1  = ab1[r * 128 + col];
    const float w2  = aW2[r * 128 + col];
#pragma unroll
    for (int v = 0; v < 8; ++v) {
      float hv = fmaxf(acc[nt][v] + b1, 0.0f);
      part[v] += hv * w2;
    }
  }
  // reduce across the 16 lanes of each half (xor masks stay within a half)
#pragma unroll
  for (int m = 1; m < 16; m <<= 1) {
#pragma unroll
    for (int v = 0; v < 8; ++v) part[v] += __shfl_xor(part[v], m, 32);
  }
  if (lr == 0) {
    const float b2 = ab2[r];
    const float em = embm[r];
#pragma unroll
    for (int v = 0; v < 8; ++v) {
      float s = part[v] + b2;
      s = (s > 0.0f) ? s : 0.2f * s;       // leaky_relu(0.2)
      s4[(size_t)(tile * 16 + 8 * lh + v) * 4 + r] = s + em;
    }
  }
}

// --------------------------- small helper kernels ---------------------------
__global__ void fill_kernel(float* p, size_t n, float v) {
  size_t i = (size_t)blockIdx.x * blockDim.x + threadIdx.x;
  if (i < n) p[i] = v;
}

__global__ void prep_kernel(const float* __restrict__ imp, const float* __restrict__ emb,
                            float* __restrict__ ew, float* __restrict__ embm) {
  int t = threadIdx.x;
  if (t < LLAYERS * RREL) {
    int l = t >> 2, r = t & 3;
    float m = imp[l * 4];
    for (int i = 1; i < 4; ++i) m = fmaxf(m, imp[l * 4 + i]);
    float s = 0.0f;
    for (int i = 0; i < 4; ++i) s += expf(imp[l * 4 + i] - m);
    ew[t] = expf(imp[l * 4 + r] - m) / s;
    float a = 0.0f;
    const float* ep = emb + (size_t)t * 64;
    for (int i = 0; i < 64; ++i) a += ep[i];
    embm[t] = a * (1.0f / 64.0f);
  }
}

__global__ void initred_kernel(float* smax, float* ssum) {
  int t = threadIdx.x;
  if (t < RREL) { smax[t] = -INFINITY; ssum[t] = 0.0f; }
}

__global__ void amax_kernel(const float* __restrict__ s4, const int* __restrict__ et,
                            float* __restrict__ smax, int E) {
  int e = blockIdx.x * blockDim.x + threadIdx.x;
  if (e >= E) return;
  int r = et[e];
  float v = s4[(size_t)e * 4 + r];
  if (v >= 0.0f) atomicMax((int*)(smax + r), __float_as_int(v));
  else           atomicMin((unsigned int*)(smax + r), (unsigned int)__float_as_int(v));
}

__global__ void sumexp_kernel(const float* __restrict__ s4, const int* __restrict__ et,
                              const float* __restrict__ smax, float* __restrict__ ssum, int E) {
  int e = blockIdx.x * blockDim.x + threadIdx.x;
  if (e >= E) return;
  int r = et[e];
  atomicAdd(ssum + r, expf(s4[(size_t)e * 4 + r] - smax[r]));
}

// agg[dst] += ew[r] * softmax_coef(e) * tx[r][src]   (one block per edge)
__global__ __launch_bounds__(256)
void scatter_kernel(const float* __restrict__ s4, const int* __restrict__ et,
                    const float* __restrict__ smax, const float* __restrict__ ssum,
                    const float* __restrict__ ew,
                    const int* __restrict__ src, const int* __restrict__ dst,
                    const float* __restrict__ tx, float* __restrict__ agg) {
  const int e = blockIdx.x;
  const int r = et[e];
  const float* txs = tx + ((size_t)r * N_NODES + src[e]) * DMODEL;
  __builtin_prefetch((const void*)(txs + threadIdx.x), 0, 3);
  const float coef = ew[r] * expf(s4[(size_t)e * 4 + r] - smax[r]) / ssum[r];
  float* ad = agg + (size_t)dst[e] * DMODEL;
  const int d = threadIdx.x;
  atomicAdd(ad + d, coef * txs[d]);
}

// out=(1-a)*agg + (h@Ws^T+bs) + a*h0 ; LayerNorm ; exact GELU ; fp32 + bf16 out
__global__ __launch_bounds__(256)
void combine_ln_gelu(const float* __restrict__ agg, const float* __restrict__ hws,
                     const float* __restrict__ h0, const float* __restrict__ g,
                     const float* __restrict__ bb, float* __restrict__ hout,
                     __bf16* __restrict__ houtb) {
  __shared__ float red[256];
  const int n = blockIdx.x, d = threadIdx.x;
  const size_t idx = (size_t)n * DMODEL + d;
  float o = (1.0f - ALPHA_C) * agg[idx] + hws[idx] + ALPHA_C * h0[idx];
  red[d] = o; __syncthreads();
  for (int s = 128; s > 0; s >>= 1) { if (d < s) red[d] += red[d + s]; __syncthreads(); }
  const float mu = red[0] * (1.0f / DMODEL); __syncthreads();
  float c = o - mu;
  red[d] = c * c; __syncthreads();
  for (int s = 128; s > 0; s >>= 1) { if (d < s) red[d] += red[d + s]; __syncthreads(); }
  const float var = red[0] * (1.0f / DMODEL);
  float y = c * rsqrtf(var + 1e-5f) * g[d] + bb[d];
  y = 0.5f * y * (1.0f + erff(y * 0.70710678118654752f));
  hout[idx]  = y;
  houtb[idx] = f2bf(y);
}

__global__ __launch_bounds__(256)
void pool_sum_kernel(const float* __restrict__ h, const int* __restrict__ batch,
                     float* __restrict__ sums, float* __restrict__ cnt) {
  const int n = blockIdx.x, d = threadIdx.x;
  const int b = batch[n];
  atomicAdd(&sums[(size_t)b * DMODEL + d], h[(size_t)n * DMODEL + d]);
  if (d == 0) atomicAdd(&cnt[b], 1.0f);
}

__global__ void pool_div_kernel(const float* __restrict__ sums, const float* __restrict__ cnt,
                                __bf16* __restrict__ pooled) {
  const int b = blockIdx.x, d = threadIdx.x;
  pooled[(size_t)b * DMODEL + d] = f2bf(sums[(size_t)b * DMODEL + d] / fmaxf(cnt[b], 1.0f));
}

// torch GRU gate math; keeps fp32 state + bf16 mirror (feeds next step's WMMA)
__global__ __launch_bounds__(256)
void gru_gate_kernel(const float* __restrict__ gi, const float* __restrict__ gh,
                     float* __restrict__ hbuf, __bf16* __restrict__ hbufb,
                     float* __restrict__ ys, int tstore) {
  const int b = blockIdx.x, d = threadIdx.x;
  const size_t o = (size_t)b * 768 + d;
  const float ir = gi[o],       hr = gh[o];
  const float iz = gi[o + 256], hz = gh[o + 256];
  const float in = gi[o + 512], hn = gh[o + 512];
  const float r  = 1.0f / (1.0f + expf(-(ir + hr)));
  const float z  = 1.0f / (1.0f + expf(-(iz + hz)));
  const float nn = tanhf(in + r * hn);
  const float hp = hbuf[(size_t)b * SHID + d];
  const float hw = (1.0f - z) * nn + z * hp;
  hbuf[(size_t)b * SHID + d]  = hw;
  hbufb[(size_t)b * SHID + d] = f2bf(hw);
  ys[((size_t)b * TSEQ + tstore) * SHID + d] = hw;
}

__global__ __launch_bounds__(512)
void seq_pool_kernel(const float* __restrict__ ysf, const float* __restrict__ ysb,
                     __bf16* __restrict__ z) {
  const int b = blockIdx.x, j = threadIdx.x;   // j < 512
  const float* base = (j < SHID) ? (ysf + (size_t)b * TSEQ * SHID + j)
                                 : (ysb + (size_t)b * TSEQ * SHID + (j - SHID));
  float s = 0.0f, mx = -INFINITY;
  for (int t = 0; t < TSEQ; ++t) { float v = base[(size_t)t * SHID]; s += v; mx = fmaxf(mx, v); }
  z[(size_t)b * 512 + j] = f2bf(s * (1.0f / TSEQ) + mx);
}

__global__ void head_kernel(const float* __restrict__ gz, const float* __restrict__ gW3,
                            const float* __restrict__ gb3, const float* __restrict__ sz,
                            const float* __restrict__ sW3, const float* __restrict__ sb3,
                            float* __restrict__ out) {
  const int b = blockIdx.x, c = threadIdx.x;
  if (c >= NCLS) return;
  float acc = gb3[c] + sb3[c];
  for (int k = 0; k < 64;  ++k) acc += gz[(size_t)b * 64  + k] * gW3[(size_t)c * 64  + k];
  for (int k = 0; k < 128; ++k) acc += sz[(size_t)b * 128 + k] * sW3[(size_t)c * 128 + k];
  out[(size_t)b * NCLS + c] = acc;
}

// ---------------------------------------------------------------------------
extern "C" void kernel_launch(void* const* d_in, const int* in_sizes, int n_in,
                              void* d_out, int out_size, void* d_ws, size_t ws_size,
                              hipStream_t stream) {
  (void)in_sizes; (void)n_in; (void)out_size; (void)ws_size;
  const float* x     = (const float*)d_in[0];
  const int*   ei    = (const int*)d_in[1];
  const int*   et    = (const int*)d_in[2];
  const int*   batch = (const int*)d_in[3];
  const float* seq   = (const float*)d_in[4];
  const float* projW = (const float*)d_in[5];
  const float* projb = (const float*)d_in[6];
  const float* Wr    = (const float*)d_in[7];
  const float* Ws    = (const float*)d_in[8];
  const float* bs    = (const float*)d_in[9];
  const float* aW1   = (const float*)d_in[10];
  const float* ab1   = (const float*)d_in[11];
  const float* aW2   = (const float*)d_in[12];
  const float* ab2   = (const float*)d_in[13];
  const float* emb   = (const float*)d_in[14];
  const float* imp   = (const float*)d_in[15];
  const float* lng   = (const float*)d_in[16];
  const float* lnb   = (const float*)d_in[17];
  const float* gmW1  = (const float*)d_in[18];
  const float* gmb1  = (const float*)d_in[19];
  const float* gmW2  = (const float*)d_in[20];
  const float* gmb2  = (const float*)d_in[21];
  const float* gmW3  = (const float*)d_in[22];
  const float* gmb3  = (const float*)d_in[23];
  const float* gWih  = (const float*)d_in[24];
  const float* gWhh  = (const float*)d_in[25];
  const float* gbih  = (const float*)d_in[26];
  const float* gbhh  = (const float*)d_in[27];
  const float* smW1  = (const float*)d_in[28];
  const float* smb1  = (const float*)d_in[29];
  const float* smW2  = (const float*)d_in[30];
  const float* smb2  = (const float*)d_in[31];
  const float* smW3  = (const float*)d_in[32];
  const float* smb3  = (const float*)d_in[33];
  float* out = (float*)d_out;

  const int* src = ei;            // edge_index[0]
  const int* dst = ei + N_EDGES;  // edge_index[1]

  char* p = (char*)d_ws;
  auto carveB = [&](size_t bytes) -> char* {
    char* q = p; p += (bytes + 255) & ~(size_t)255; return q;
  };
  auto carveF  = [&](size_t n) { return (float*)carveB(n * 4); };
  auto carveH  = [&](size_t n) { return (__bf16*)carveB(n * 2); };

  float*  h     = carveF((size_t)N_NODES * DMODEL);
  float*  h0    = carveF((size_t)N_NODES * DMODEL);
  float*  tmp   = carveF((size_t)N_NODES * DMODEL);
  float*  agg   = carveF((size_t)N_NODES * DMODEL);
  float*  tx    = carveF((size_t)RREL * N_NODES * DMODEL);
  float*  s4    = carveF((size_t)N_EDGES * 4);
  float*  smax  = carveF(16);
  float*  ssum  = carveF(16);
  float*  ew    = carveF(16);
  float*  embm  = carveF(16);
  float*  sums  = carveF((size_t)BGRAPH * DMODEL);
  float*  cnt   = carveF(BGRAPH);
  float*  gi    = carveF((size_t)BGRAPH * 768);
  float*  gh    = carveF((size_t)BGRAPH * 768);
  float*  hseq  = carveF((size_t)BGRAPH * SHID);
  float*  ysf   = carveF((size_t)BGRAPH * TSEQ * SHID);
  float*  ysb   = carveF((size_t)BGRAPH * TSEQ * SHID);
  float*  z2    = carveF((size_t)BGRAPH * 128);
  float*  gz2   = carveF((size_t)BGRAPH * 64);
  // bf16 operands
  __bf16* xbf    = carveH((size_t)N_NODES * DIN);
  __bf16* hbf    = carveH((size_t)N_NODES * DMODEL);
  __bf16* seqbf  = carveH((size_t)BGRAPH * TSEQ * SDIM);
  __bf16* hseqbf = carveH((size_t)BGRAPH * SHID);
  __bf16* zzb    = carveH((size_t)BGRAPH * 512);
  __bf16* z1b    = carveH((size_t)BGRAPH * 256);
  __bf16* pooledb= carveH((size_t)BGRAPH * DMODEL);
  __bf16* gz1b   = carveH((size_t)BGRAPH * 128);
  __bf16* projWb = carveH((size_t)DMODEL * DIN);
  __bf16* Wrb    = carveH((size_t)LLAYERS * RREL * DMODEL * DMODEL);
  __bf16* Wsb    = carveH((size_t)LLAYERS * DMODEL * DMODEL);
  __bf16* aW1b   = carveH((size_t)LLAYERS * RREL * 128 * 512);
  __bf16* gWihb  = carveH((size_t)2 * 768 * SDIM);
  __bf16* gWhhb  = carveH((size_t)2 * 768 * SHID);
  __bf16* smW1b  = carveH((size_t)SHID * 512);
  __bf16* smW2b  = carveH((size_t)128 * SHID);
  __bf16* gmW1b  = carveH((size_t)128 * DMODEL);
  __bf16* gmW2b  = carveH((size_t)64 * 128);

  auto gemm = [&](const __bf16* A, int lda, const __bf16* W, int ldw, const float* bias,
                  float* Cf, __bf16* Cb, int ldc, int M, int N, int K, int act) {
    int tiles  = (M >> 4) * (N >> 5);       // 16x32 tiles per wave
    int blocks = (tiles + 7) / 8;
    gemm_bf16_wmma<<<blocks, 256, 0, stream>>>(A, lda, W, ldw, bias, Cf, Cb, ldc, M, N, K, act);
  };
  auto fill = [&](float* ptr, size_t n, float v) {
    fill_kernel<<<(unsigned)((n + 255) / 256), 256, 0, stream>>>(ptr, n, v);
  };
  auto conv = [&](const float* in, __bf16* o, size_t n) {
    f32_to_bf16_kernel<<<(unsigned)((n + 255) / 256), 256, 0, stream>>>(in, o, n);
  };

  // ------------------- one-shot bf16 operand conversion --------------------
  conv(x,     xbf,    (size_t)N_NODES * DIN);
  conv(seq,   seqbf,  (size_t)BGRAPH * TSEQ * SDIM);
  conv(projW, projWb, (size_t)DMODEL * DIN);
  conv(Wr,    Wrb,    (size_t)LLAYERS * RREL * DMODEL * DMODEL);
  conv(Ws,    Wsb,    (size_t)LLAYERS * DMODEL * DMODEL);
  conv(aW1,   aW1b,   (size_t)LLAYERS * RREL * 128 * 512);
  conv(gWih,  gWihb,  (size_t)2 * 768 * SDIM);
  conv(gWhh,  gWhhb,  (size_t)2 * 768 * SHID);
  conv(smW1,  smW1b,  (size_t)SHID * 512);
  conv(smW2,  smW2b,  (size_t)128 * SHID);
  conv(gmW1,  gmW1b,  (size_t)128 * DMODEL);
  conv(gmW2,  gmW2b,  (size_t)64 * 128);

  // ------------------ sequence branch (bidirectional GRU) ------------------
  fill(hseq, (size_t)BGRAPH * SHID, 0.0f);
  fill((float*)hseqbf, (size_t)BGRAPH * SHID / 2, 0.0f);   // bf16 zeros = zero bits
  for (int t = 0; t < TSEQ; ++t) {
    gemm(seqbf + (size_t)t * SDIM, TSEQ * SDIM, gWihb, SDIM, gbih, gi, nullptr, 3 * SHID,
         BGRAPH, 3 * SHID, SDIM, 0);
    gemm(hseqbf, SHID, gWhhb, SHID, gbhh, gh, nullptr, 3 * SHID, BGRAPH, 3 * SHID, SHID, 0);
    gru_gate_kernel<<<BGRAPH, SHID, 0, stream>>>(gi, gh, hseq, hseqbf, ysf, t);
  }
  fill(hseq, (size_t)BGRAPH * SHID, 0.0f);
  fill((float*)hseqbf, (size_t)BGRAPH * SHID / 2, 0.0f);
  for (int t = 0; t < TSEQ; ++t) {
    int ti = TSEQ - 1 - t;  // reversed input; store back at original index
    gemm(seqbf + (size_t)ti * SDIM, TSEQ * SDIM, gWihb + (size_t)3 * SHID * SDIM, SDIM,
         gbih + 3 * SHID, gi, nullptr, 3 * SHID, BGRAPH, 3 * SHID, SDIM, 0);
    gemm(hseqbf, SHID, gWhhb + (size_t)3 * SHID * SHID, SHID, gbhh + 3 * SHID, gh, nullptr,
         3 * SHID, BGRAPH, 3 * SHID, SHID, 0);
    gru_gate_kernel<<<BGRAPH, SHID, 0, stream>>>(gi, gh, hseq, hseqbf, ysb, ti);
  }
  seq_pool_kernel<<<BGRAPH, 2 * SHID, 0, stream>>>(ysf, ysb, zzb);
  gemm(zzb, 2 * SHID, smW1b, 2 * SHID, smb1, nullptr, z1b, SHID, BGRAPH, SHID, 2 * SHID, 1);
  gemm(z1b, SHID, smW2b, SHID, smb2, z2, nullptr, 128, BGRAPH, 128, SHID, 1);

  // ----------------------------- graph branch -----------------------------
  prep_kernel<<<1, 64, 0, stream>>>(imp, emb, ew, embm);
  gemm(xbf, DIN, projWb, DIN, projb, h, hbf, DMODEL, N_NODES, DMODEL, DIN, 0);
  hipMemcpyAsync(h0, h, (size_t)N_NODES * DMODEL * 4, hipMemcpyDeviceToDevice, stream);

  for (int l = 0; l < LLAYERS; ++l) {
    initred_kernel<<<1, 32, 0, stream>>>(smax, ssum);
    edge_score_wmma<<<(N_EDGES / 16) * RREL / 8, 256, 0, stream>>>(
        hbf, src, dst,
        aW1b + (size_t)l * RREL * 128 * 512, ab1 + (size_t)l * RREL * 128,
        aW2 + (size_t)l * RREL * 128, ab2 + (size_t)l * RREL,
        embm + l * RREL, s4);
    amax_kernel<<<(N_EDGES + 255) / 256, 256, 0, stream>>>(s4, et, smax, N_EDGES);
    sumexp_kernel<<<(N_EDGES + 255) / 256, 256, 0, stream>>>(s4, et, smax, ssum, N_EDGES);
    for (int r = 0; r < RREL; ++r)
      gemm(hbf, DMODEL, Wrb + (size_t)(l * RREL + r) * DMODEL * DMODEL, DMODEL, nullptr,
           tx + (size_t)r * N_NODES * DMODEL, nullptr, DMODEL, N_NODES, DMODEL, DMODEL, 0);
    fill(agg, (size_t)N_NODES * DMODEL, 0.0f);
    scatter_kernel<<<N_EDGES, DMODEL, 0, stream>>>(s4, et, smax, ssum, ew + l * RREL,
                                                   src, dst, tx, agg);
    gemm(hbf, DMODEL, Wsb + (size_t)l * DMODEL * DMODEL, DMODEL, bs + (size_t)l * DMODEL,
         tmp, nullptr, DMODEL, N_NODES, DMODEL, DMODEL, 0);
    combine_ln_gelu<<<N_NODES, DMODEL, 0, stream>>>(agg, tmp, h0,
                                                    lng + (size_t)l * DMODEL,
                                                    lnb + (size_t)l * DMODEL, h, hbf);
  }

  fill(sums, (size_t)BGRAPH * DMODEL, 0.0f);
  fill(cnt, BGRAPH, 0.0f);
  pool_sum_kernel<<<N_NODES, DMODEL, 0, stream>>>(h, batch, sums, cnt);
  pool_div_kernel<<<BGRAPH, DMODEL, 0, stream>>>(sums, cnt, pooledb);
  gemm(pooledb, DMODEL, gmW1b, DMODEL, gmb1, nullptr, gz1b, 128, BGRAPH, 128, DMODEL, 1);
  gemm(gz1b, 128, gmW2b, 128, gmb2, gz2, nullptr, 64, BGRAPH, 64, 128, 1);
  head_kernel<<<BGRAPH, 32, 0, stream>>>(gz2, gmW3, gmb3, z2, smW3, smb3, out);
}
